// LEBNN_40149354283211
// MI455X (gfx1250) — matrix operand order repaired and checked by reading, not compiled, not run on hardware
//
#include <hip/hip_runtime.h>

// ---------------------------------------------------------------------------
// LEBNN gradient, fused:  grad = u*(1 + g*Σβ) − g*(β·C)
//   GEMM1 (Gᵀ = C·Uᵀ) in split bf16 (hi·hi + hi·lo + lo·hi) → fp32-grade φ
//   GEMM2 (β·C) in bf16; all elementwise + reductions in fp32.
//   C operands double-buffered in LDS via async global->LDS (ASYNCcnt).
// ---------------------------------------------------------------------------

typedef __attribute__((ext_vector_type(16))) __bf16       v16bf;
typedef __attribute__((ext_vector_type(8)))  float        v8f;
typedef __attribute__((ext_vector_type(4)))  unsigned int u32x4;
typedef __attribute__((ext_vector_type(4)))  int          v4i;

typedef __attribute__((address_space(1))) v4i as1_v4i;   // global int4
typedef __attribute__((address_space(3))) v4i as3_v4i;   // LDS int4

union V16 { v16bf v; u32x4 q[2]; };

#if defined(__has_builtin)
#if __has_builtin(__builtin_amdgcn_global_load_async_to_lds_b128)
#define LEBNN_ASYNC 1
#endif
#endif

static __device__ __forceinline__ v8f wmma_bf16(v16bf a, v16bf b, v8f c) {
  return __builtin_amdgcn_wmma_f32_16x16x32_bf16(false, a, false, b, (short)0, c,
                                                 false, false);
}

#ifdef LEBNN_ASYNC
static __device__ __forceinline__ void async_g2l_b128(const void* g, void* l) {
  __builtin_amdgcn_global_load_async_to_lds_b128(
      (as1_v4i*)g, (as3_v4i*)l, 0, 0);
}
static __device__ __forceinline__ void wait_async0() {
#if __has_builtin(__builtin_amdgcn_s_wait_asynccnt)
  __builtin_amdgcn_s_wait_asynccnt(0);
#else
  asm volatile("s_wait_asynccnt 0x0" ::: "memory");
#endif
}
#endif

// ---- preprocessing: bf16 split of C, transposed C, per-m parameters --------
__global__ __launch_bounds__(128) void lebnn_prep(
    const float* __restrict__ C, const float* __restrict__ log_sig,
    const float* __restrict__ w,
    __bf16* __restrict__ Chi, __bf16* __restrict__ Clo,
    __bf16* __restrict__ CT, float4* __restrict__ params)
{
  const int m = blockIdx.x;   // 0..1023
  const int l = threadIdx.x;  // 0..127
  float c   = C[m * 128 + l];
  __bf16 hi = (__bf16)c;
  __bf16 lo = (__bf16)(c - (float)hi);
  Chi[m * 128 + l] = hi;
  Clo[m * 128 + l] = lo;
  CT[(size_t)l * 1024 + m] = hi;

  __shared__ float red[128];
  red[l] = c * c;
  __syncthreads();
  for (int s = 64; s > 0; s >>= 1) {
    if (l < s) red[l] += red[l + s];
    __syncthreads();
  }
  if (l == 0) {
    float csq   = red[0];
    float isig2 = expf(-2.0f * log_sig[m]);   // 1/sigma^2
    float wm    = w[m];
    params[m]   = make_float4(csq, isig2, wm, wm * isig2);
  }
}

// ---- main fused kernel: 1 WG = 64 rows of B (4 waves x 16 rows) ------------
__global__ __launch_bounds__(128) void lebnn_main(
    const float*  __restrict__ U,
    const __bf16* __restrict__ Chi, const __bf16* __restrict__ Clo,
    const __bf16* __restrict__ CT,  const float4* __restrict__ params,
    float* __restrict__ out)
{
  constexpr int L = 128, M = 1024;
  constexpr int CH_STRIDE = 272;  // 256B row + 16B pad (bank shift, 16B aligned)
  constexpr int CT_STRIDE = 80;   // 64B row + 16B pad
  constexpr int SB = 32 * CH_STRIDE * 2 + 128 * CT_STRIDE;  // 27648 B / buffer
  __shared__ __align__(16) char smem[2 * SB];

  const int tid  = threadIdx.x;
  const int wave = tid >> 5;
  const int lane = tid & 31;
  const int n    = lane & 15;   // b-column this lane owns in D tiles
  const int h    = lane >> 4;   // lane-half
  const int row0 = blockIdx.x * 64 + wave * 16;

  // stage C_hi/C_lo rows m0..m0+31 and CT cols m0..m0+31 into buffer p
  auto stage = [&](int mc, int p) {
    char* base = smem + p * SB;
    const int m0 = mc * 32;
    const int rrow = tid >> 2, seg = tid & 3;
    const char* gh = (const char*)(Chi + (size_t)(m0 + rrow) * L + seg * 32);
    const char* gl = (const char*)(Clo + (size_t)(m0 + rrow) * L + seg * 32);
    const char* gt = (const char*)(CT + (size_t)tid * M + m0);
    char* dh = base + rrow * CH_STRIDE + seg * 64;
    char* dl = base + 32 * CH_STRIDE + rrow * CH_STRIDE + seg * 64;
    char* dt = base + 64 * CH_STRIDE + tid * CT_STRIDE;
#ifdef LEBNN_ASYNC
#pragma unroll
    for (int i = 0; i < 4; ++i) {
      async_g2l_b128(gh + i * 16, dh + i * 16);
      async_g2l_b128(gl + i * 16, dl + i * 16);
      async_g2l_b128(gt + i * 16, dt + i * 16);
    }
#else
    u32x4 rh[4], rl[4], rt[4];
#pragma unroll
    for (int i = 0; i < 4; ++i) {
      rh[i] = *(const u32x4*)(gh + i * 16);
      rl[i] = *(const u32x4*)(gl + i * 16);
      rt[i] = *(const u32x4*)(gt + i * 16);
    }
#pragma unroll
    for (int i = 0; i < 4; ++i) {
      *(u32x4*)(dh + i * 16) = rh[i];
      *(u32x4*)(dl + i * 16) = rl[i];
      *(u32x4*)(dt + i * 16) = rt[i];
    }
#endif
  };

  // -- load this lane's U row, split to bf16 hi/lo in B-operand layout --
  // B-operand (U as 32x16 per chunk): lane-half h holds k 8h..8h+7, 16+8h..16+8h+7.
  V16 ahi[4], alo[4];
  float usq_p = 0.f;
  const float* urow = U + (size_t)(row0 + n) * L;
#pragma unroll
  for (int c = 0; c < 4; ++c) {
    const float4* p = (const float4*)(urow + c * 32 + h * 8);
    float4 f4[4] = {p[0], p[1], p[4], p[5]};
    const float* f = (const float*)f4;
#pragma unroll
    for (int i = 0; i < 16; ++i) {
      float  x  = f[i];
      __bf16 hi = (__bf16)x;
      __bf16 lo = (__bf16)(x - (float)hi);
      ahi[c].v[i] = hi;
      alo[c].v[i] = lo;
      usq_p = fmaf(x, x, usq_p);
    }
  }
  const float usq = usq_p + __shfl_xor(usq_p, 16, 32);  // ||u_{row0+n}||^2

  const v8f vzero = {0.f, 0.f, 0.f, 0.f, 0.f, 0.f, 0.f, 0.f};
  v8f oacc[8];
#pragma unroll
  for (int i = 0; i < 8; ++i) oacc[i] = vzero;
  float s_acc = 0.f, brs_acc = 0.f;

  stage(0, 0);  // prologue fill of buffer 0

  for (int mc = 0; mc < 32; ++mc) {
    const int p  = mc & 1;
    const int m0 = mc * 32;
#ifdef LEBNN_ASYNC
    wait_async0();           // this wave's writes for chunk mc have landed in LDS
#endif
    __syncthreads();         // all waves landed; all done reading buffer 1-p
    if (mc < 31) stage(mc + 1, 1 - p);  // overlap next chunk's fill with compute

    const char* sChi = smem + p * SB;
    const char* sClo = sChi + 32 * CH_STRIDE;
    const char* sCT  = sClo + 32 * CH_STRIDE;

    // Gᵀ tiles for m0..m0+31: D lands exactly in GEMM2's A-operand packing.
    V16 abeta;
#pragma unroll
    for (int t = 0; t < 2; ++t) {
      v8f gacc = vzero;
#pragma unroll
      for (int c = 0; c < 4; ++c) {  // K = 128 in chunks of 32
        V16 bhi, blo;
        const char* rp = sChi + (t * 16 + n) * CH_STRIDE + c * 64 + h * 16;
        const char* rl = sClo + (t * 16 + n) * CH_STRIDE + c * 64 + h * 16;
        bhi.q[0] = *(const u32x4*)rp;
        bhi.q[1] = *(const u32x4*)(rp + 32);
        blo.q[0] = *(const u32x4*)rl;
        blo.q[1] = *(const u32x4*)(rl + 32);
        gacc = wmma_bf16(bhi.v, ahi[c].v, gacc);  // hi*hi
        gacc = wmma_bf16(bhi.v, alo[c].v, gacc);  // hi*lo
        gacc = wmma_bf16(blo.v, ahi[c].v, gacc);  // lo*hi
      }
#pragma unroll
      for (int r = 0; r < 8; ++r) {
        const int m = m0 + t * 16 + h * 8 + r;    // m-row held in gacc[r]
        const float4 pm = params[m];              // csq, 1/s^2, w, w/s^2
        float G  = gacc[r];
        float tt = usq + pm.x - 2.0f * G;         // ||u-c||^2
        float tc = fmaxf(tt, 0.0f);
        float x  = fmaf(tc, pm.y, 1.0f);          // phi^2
        float rx = __frsqrt_rn(x);                // 1/phi
        s_acc = fmaf(x * rx, pm.z, s_acc);        // += phi*w
        float beta = (tt > 0.0f) ? pm.w * rx : 0.0f;
        brs_acc += beta;
        abeta.v[t * 8 + r] = (__bf16)beta;        // lands in A-operand slot
      }
    }
    // GEMM2 partial: out += beta(16x32) * C[m0..m0+31][:]
#pragma unroll
    for (int lt = 0; lt < 8; ++lt) {
      V16 bt;
      const char* rp = sCT + (lt * 16 + n) * CT_STRIDE + h * 16;
      bt.q[0] = *(const u32x4*)rp;
      bt.q[1] = *(const u32x4*)(rp + 32);
      oacc[lt] = wmma_bf16(abeta.v, bt.v, oacc[lt]);
    }
  }

  // per-row scalars: s, Sum(beta) -> g, fac; redistribute col->row layout
  const float s_tot   = s_acc   + __shfl_xor(s_acc, 16, 32);
  const float brs_tot = brs_acc + __shfl_xor(brs_acc, 16, 32);
  const float g   = (s_tot > 0.0f) ? 1.0f : expf(s_tot);
  const float fac = fmaf(g, brs_tot, 1.0f);
  float gv[8], fv[8];
#pragma unroll
  for (int r = 0; r < 8; ++r) {
    gv[r] = __shfl(g,   h * 8 + r, 32);
    fv[r] = __shfl(fac, h * 8 + r, 32);
  }

  const float* ur0 = U   + (size_t)row0 * L;
  float*       or0 = out + (size_t)row0 * L;
#pragma unroll
  for (int lt = 0; lt < 8; ++lt) {
#pragma unroll
    for (int r = 0; r < 8; ++r) {
      const int b = h * 8 + r;
      const int l = lt * 16 + n;
      float uv = ur0[b * L + l];
      or0[b * L + l] = fmaf(uv, fv[r], -gv[r] * oacc[lt][r]);
    }
  }
}

// ---------------------------------------------------------------------------
extern "C" void kernel_launch(void* const* d_in, const int* in_sizes, int n_in,
                              void* d_out, int out_size, void* d_ws, size_t ws_size,
                              hipStream_t stream) {
  (void)in_sizes; (void)n_in; (void)out_size; (void)ws_size;
  const float* U    = (const float*)d_in[0];  // [32768,128]
  const float* C    = (const float*)d_in[1];  // [1024,128]
  const float* logs = (const float*)d_in[2];  // [1024]
  const float* w    = (const float*)d_in[3];  // [1024]
  float* out = (float*)d_out;                 // [32768,128]

  char* ws = (char*)d_ws;
  __bf16* Chi    = (__bf16*)(ws);                //  256 KB
  __bf16* Clo    = (__bf16*)(ws + (256 << 10));  //  256 KB
  __bf16* CT     = (__bf16*)(ws + (512 << 10));  //  256 KB
  float4* params = (float4*)(ws + (768 << 10));  //   16 KB

  lebnn_prep<<<dim3(1024), dim3(128), 0, stream>>>(C, logs, w, Chi, Clo, CT, params);
  lebnn_main<<<dim3(512),  dim3(128), 0, stream>>>(U, Chi, Clo, CT, params, out);
}
